// BinarizedRBF_30511447670912
// MI455X (gfx1250) — matrix operand order repaired and checked by reading, not compiled, not run on hardware
//
#include <hip/hip_runtime.h>
#include <hip/hip_bf16.h>

#define BN 4096
#define CN 4096
#define DN 2048
#define GAMMA_F 0.05f

typedef __attribute__((ext_vector_type(16))) __bf16 v16bf;
typedef __attribute__((ext_vector_type(8)))  __bf16 v8bf;
typedef __attribute__((ext_vector_type(4)))  __bf16 v4bf;
typedef __attribute__((ext_vector_type(8)))  float  v8f;
typedef __attribute__((ext_vector_type(4)))  int    v4i;

#define AS_GLOBAL __attribute__((address_space(1)))
#define AS_LDS    __attribute__((address_space(3)))

#if defined(__has_builtin)
#  if __has_builtin(__builtin_amdgcn_global_load_async_to_lds_b128) && \
      __has_builtin(__builtin_amdgcn_s_wait_asynccnt)
#    define USE_ASYNC 1
#  endif
#endif
#ifndef USE_ASYNC
#  define USE_ASYNC 0
#endif

// fp32 -> bf16 hi + bf16 lo (RNE), so hi*hi' + lo*hi' + hi*lo' ~ fp32 product.
__device__ __forceinline__ void bf_split(float f, __bf16 &hi, __bf16 &lo) {
  unsigned u  = __builtin_bit_cast(unsigned, f);
  unsigned rh = (u + 0x7FFFu + ((u >> 16) & 1u)) & 0xFFFF0000u;
  hi = __builtin_bit_cast(__bf16, (unsigned short)(rh >> 16));
  float fl = f - __builtin_bit_cast(float, rh);
  unsigned ul = __builtin_bit_cast(unsigned, fl);
  lo = __builtin_bit_cast(__bf16,
        (unsigned short)((ul + 0x7FFFu + ((ul >> 16) & 1u)) >> 16));
}

// ---------------------------------------------------------------------------
// Pre-pass: one block per row. Splits fp32 row into bf16 hi/lo planes and
// computes the squared norm in the same read.
// ---------------------------------------------------------------------------
__global__ __launch_bounds__(256)
void split_norm_kernel(const float* __restrict__ in,
                       __bf16* __restrict__ hip_,
                       __bf16* __restrict__ lop_,
                       float* __restrict__ sq, int cols) {
  __shared__ float red[256];
  const int row = blockIdx.x;
  const float* p = in + (size_t)row * cols;
  __bf16* ph = hip_ + (size_t)row * cols;
  __bf16* pl = lop_ + (size_t)row * cols;
  float s = 0.0f;
  for (int i = threadIdx.x * 4; i < cols; i += blockDim.x * 4) {
    float4 v = *(const float4*)(p + i);
    v4bf h4, l4;
    __bf16 hb, lb;
    bf_split(v.x, hb, lb); h4[0] = hb; l4[0] = lb;
    bf_split(v.y, hb, lb); h4[1] = hb; l4[1] = lb;
    bf_split(v.z, hb, lb); h4[2] = hb; l4[2] = lb;
    bf_split(v.w, hb, lb); h4[3] = hb; l4[3] = lb;
    *(v4bf*)(ph + i) = h4;
    *(v4bf*)(pl + i) = l4;
    s += v.x * v.x + v.y * v.y + v.z * v.z + v.w * v.w;
  }
  red[threadIdx.x] = s;
  __syncthreads();
  for (int off = 128; off > 0; off >>= 1) {
    if (threadIdx.x < off) red[threadIdx.x] += red[threadIdx.x + off];
    __syncthreads();
  }
  if (threadIdx.x == 0) sq[row] = red[0];
}

// ---------------------------------------------------------------------------
// Fused RBF kernel: 128x128 output tile per 128-thread block (4 waves, 2x2
// wave grid, 64x64 per wave -> wmma:ds_load = 48:32 per K-chunk). bf16x3
// WMMA GEMM over pre-split bf16 planes, async global->LDS double buffering,
// exp epilogue.
// ---------------------------------------------------------------------------
#define KC 32            // K chunk per iteration (bf16 elems)
#define LW 40            // padded LDS row stride in bf16 (80B, 16B multiple)

__global__ __launch_bounds__(128)
void rbf_wmma_kernel(const __bf16* __restrict__ xhg,
                     const __bf16* __restrict__ xlg,
                     const __bf16* __restrict__ chg,
                     const __bf16* __restrict__ clg,
                     const float* __restrict__ xsq,
                     const float* __restrict__ csq,
                     float* __restrict__ out) {
  __shared__ __align__(16) __bf16 xh[2][128][LW];
  __shared__ __align__(16) __bf16 xl[2][128][LW];
  __shared__ __align__(16) __bf16 ch[2][128][LW];
  __shared__ __align__(16) __bf16 cl[2][128][LW];

  const int tid    = threadIdx.x;
  const int lane   = tid & 31;
  const int wid    = tid >> 5;        // 0..3
  const int wave_m = wid & 1;         // 2 waves along M (64 rows each)
  const int wave_n = wid >> 1;        // 2 waves along N (64 cols each)
  const int h      = lane >> 4;       // lane half (0/1)
  const int mr     = lane & 15;       // row/col within 16
  const int rowBlock = blockIdx.y * 128;
  const int colBlock = blockIdx.x * 128;

  // Stage one K-chunk (128 x 32 bf16 of each of the 4 planes) into LDS.
  auto copy_tiles = [&](int buf, int kc) {
    const int kcol = kc * KC;
#pragma unroll
    for (int i = 0; i < 4; ++i) {
      int lin = i * 128 + tid;        // 512 transfers of 16B per plane
      int r   = lin >> 2;             // 0..127
      int seg = (lin & 3) * 8;        // bf16 column, 16B granules
      size_t gx = (size_t)(rowBlock + r) * DN + kcol + seg;
      size_t gc = (size_t)(colBlock + r) * DN + kcol + seg;
#if USE_ASYNC
      __builtin_amdgcn_global_load_async_to_lds_b128(
          (AS_GLOBAL v4i*)(xhg + gx), (AS_LDS v4i*)&xh[buf][r][seg], 0, 0);
      __builtin_amdgcn_global_load_async_to_lds_b128(
          (AS_GLOBAL v4i*)(xlg + gx), (AS_LDS v4i*)&xl[buf][r][seg], 0, 0);
      __builtin_amdgcn_global_load_async_to_lds_b128(
          (AS_GLOBAL v4i*)(chg + gc), (AS_LDS v4i*)&ch[buf][r][seg], 0, 0);
      __builtin_amdgcn_global_load_async_to_lds_b128(
          (AS_GLOBAL v4i*)(clg + gc), (AS_LDS v4i*)&cl[buf][r][seg], 0, 0);
#else
      *(v8bf*)&xh[buf][r][seg] = *(const v8bf*)(xhg + gx);
      *(v8bf*)&xl[buf][r][seg] = *(const v8bf*)(xlg + gx);
      *(v8bf*)&ch[buf][r][seg] = *(const v8bf*)(chg + gc);
      *(v8bf*)&cl[buf][r][seg] = *(const v8bf*)(clg + gc);
#endif
    }
  };

  // Two contiguous 8x-bf16 LDS vectors -> one 16-elem WMMA fragment.
  auto ld16 = [](const __bf16* p0, const __bf16* p1) -> v16bf {
    v8bf a = *(const v8bf*)p0;
    v8bf b = *(const v8bf*)p1;
    return __builtin_shufflevector(a, b, 0, 1, 2, 3, 4, 5, 6, 7,
                                   8, 9, 10, 11, 12, 13, 14, 15);
  };

  v8f acc[4][4] = {};

  copy_tiles(0, 0);
  const int NK = DN / KC;             // 64 chunks
#pragma unroll 1
  for (int kc = 0; kc < NK; ++kc) {
    const int cur = kc & 1;
#if USE_ASYNC
    __builtin_amdgcn_s_wait_asynccnt(0);
#endif
    __syncthreads();
    if (kc + 1 < NK) copy_tiles(cur ^ 1, kc + 1);

    // B fragments: lane half h covers K = h*16 .. h*16+15 contiguously.
    v16bf bhi[4], blo[4];
#pragma unroll
    for (int nt = 0; nt < 4; ++nt) {
      const int rn = wave_n * 64 + nt * 16 + mr;
      bhi[nt] = ld16(&ch[cur][rn][h * 16], &ch[cur][rn][h * 16 + 8]);
      blo[nt] = ld16(&cl[cur][rn][h * 16], &cl[cur][rn][h * 16 + 8]);
    }

#pragma unroll
    for (int mt = 0; mt < 4; ++mt) {
      // A fragment: lane half h covers K = h*16 .. h*16+15 contiguously.
      const int rm = wave_m * 64 + mt * 16 + mr;
      v16bf ahi = ld16(&xh[cur][rm][h * 16], &xh[cur][rm][h * 16 + 8]);
      v16bf alo = ld16(&xl[cur][rm][h * 16], &xl[cur][rm][h * 16 + 8]);
#pragma unroll
      for (int nt = 0; nt < 4; ++nt) {
        acc[mt][nt] = __builtin_amdgcn_wmma_f32_16x16x32_bf16(
            false, ahi, false, bhi[nt], (short)0, acc[mt][nt], false, false);
        acc[mt][nt] = __builtin_amdgcn_wmma_f32_16x16x32_bf16(
            false, alo, false, bhi[nt], (short)0, acc[mt][nt], false, false);
        acc[mt][nt] = __builtin_amdgcn_wmma_f32_16x16x32_bf16(
            false, ahi, false, blo[nt], (short)0, acc[mt][nt], false, false);
      }
    }
  }

  // Epilogue: dist = ||x||^2 + ||c||^2 - 2*cross ; out = exp(-gamma*dist).
  // C/D layout: VGPR r -> row (mt*16 + 8*h + r), lane mr -> col.
#pragma unroll
  for (int mt = 0; mt < 4; ++mt) {
    const int row0 = rowBlock + wave_m * 64 + mt * 16 + h * 8;
#pragma unroll
    for (int nt = 0; nt < 4; ++nt) {
      const int col = colBlock + wave_n * 64 + nt * 16 + mr;
      const float cn = csq[col];
#pragma unroll
      for (int r = 0; r < 8; ++r) {
        const float d = xsq[row0 + r] + cn - 2.0f * acc[mt][nt][r];
        out[(size_t)(row0 + r) * CN + col] = __expf(-GAMMA_F * d);
      }
    }
  }
}

extern "C" void kernel_launch(void* const* d_in, const int* in_sizes, int n_in,
                              void* d_out, int out_size, void* d_ws, size_t ws_size,
                              hipStream_t stream) {
  const float* x    = (const float*)d_in[0];   // (4096, 2048) f32
  const float* cent = (const float*)d_in[1];   // (4096, 2048) f32
  float* out = (float*)d_out;                  // (4096, 4096) f32

  // Workspace layout: norms (32 KB) + four bf16 planes (16 MB each).
  float*  xsq = (float*)d_ws;
  float*  csq = xsq + BN;
  __bf16* xhg = (__bf16*)(csq + CN);
  __bf16* xlg = xhg + (size_t)BN * DN;
  __bf16* chg = xlg + (size_t)BN * DN;
  __bf16* clg = chg + (size_t)CN * DN;

  split_norm_kernel<<<BN, 256, 0, stream>>>(x, xhg, xlg, xsq, DN);
  split_norm_kernel<<<CN, 256, 0, stream>>>(cent, chg, clg, csq, DN);

  dim3 grid(CN / 128, BN / 128);
  rbf_wmma_kernel<<<grid, 128, 0, stream>>>(xhg, xlg, chg, clg, xsq, csq, out);
}
